// PolicyLSTM_66949950210665
// MI455X (gfx1250) — matrix operand order repaired and checked by reading, not compiled, not run on hardware
//
#include <hip/hip_runtime.h>

typedef __attribute__((ext_vector_type(16))) _Float16 v16h;
typedef __attribute__((ext_vector_type(8)))  float    v8f;

#define T_LEN    512
#define H_DIM    5
#define G_DIM    20          // 4*H
#define IN_DIM   12
#define WG_BATCH 64          // batch rows per workgroup (== blockDim.x, 1 lane per row)
#define TC       4           // timesteps per chunk
#define NCHUNK   (T_LEN / TC)
#define ROWS     (WG_BATCH * TC)   // 256 (b,t) rows per chunk
#define M_TILES  (ROWS / 16)       // 16
#define XSTRIDE  16          // halfs per LDS x row (12 data + 4 zero pad -> 32B aligned)

__device__ __forceinline__ float sigm(float x) {
    // 1 / (1 + 2^(-x*log2e))  -> v_exp_f32 + v_rcp_f32, safe at both infinities
    return __builtin_amdgcn_rcpf(1.0f + __builtin_amdgcn_exp2f(-1.44269504f * x));
}

__device__ __forceinline__ float fast_tanh(float x) {
#if __has_builtin(__builtin_amdgcn_tanhf)
    return __builtin_amdgcn_tanhf(x);
#else
    const float ax = __builtin_fabsf(x);
    const float e  = __builtin_amdgcn_exp2f(-2.88539008f * ax);   // 2^(-2|x|*log2e)
    const float t  = (1.0f - e) * __builtin_amdgcn_rcpf(1.0f + e);
    return __builtin_copysignf(t, x);
#endif
}

__global__ __launch_bounds__(WG_BATCH, 2)
void policy_lstm_fused(const float* __restrict__ obs,   // [B,T,8]
                       const float* __restrict__ act,   // [B,T,4]
                       const float* __restrict__ W_ih,  // [20,12]
                       const float* __restrict__ W_hh,  // [20,5]
                       const float* __restrict__ b_ih,  // [20]
                       const float* __restrict__ b_hh,  // [20]
                       const float* __restrict__ W_fc,  // [4,5]
                       const float* __restrict__ b_fc,  // [4]
                       float* __restrict__ out)         // [B,4]
{
    __shared__ __align__(16) _Float16 xlds[ROWS * XSTRIDE];   // 8 KB
    __shared__ __align__(16) float    gxlds[ROWS * G_DIM];    // 20 KB

    const int tid  = threadIdx.x;          // 0..63, one batch row per lane
    const int lane = tid & 31;
    const int wave = tid >> 5;
    const int hi   = lane >> 4;            // 0: K=0..15, 1: K=16..31 zero pad (A frag)
    const int l15  = lane & 15;
    const int b    = blockIdx.x * WG_BATCH + tid;

    // ---- per-lane recurrent / head weights in VGPRs (loaded once) ----
    float whh[G_DIM * H_DIM];              // whh[g*5+k] = W_hh[g][k]
#pragma unroll
    for (int q = 0; q < 25; ++q) {
        const float4 v = ((const float4*)W_hh)[q];
        whh[4*q+0] = v.x; whh[4*q+1] = v.y; whh[4*q+2] = v.z; whh[4*q+3] = v.w;
    }
    float wfc[20];
#pragma unroll
    for (int q = 0; q < 20; ++q) wfc[q] = W_fc[q];
    float bfc[4];
#pragma unroll
    for (int q = 0; q < 4; ++q) bfc[q] = b_fc[q];

    // ---- WMMA B fragments: B[k][n] = W_ih[n][k], two N-tiles (N=0..15, 16..19) ----
    // B uses contiguous k (as A): lanes 0-15 hold K=0..15 (halves j <-> K=j), lanes 16-31 K=16..31 (zero pad)
    v16h bfr[2];
#pragma unroll
    for (int nt = 0; nt < 2; ++nt) {
        v16h bv = {};
        const int n = nt * 16 + lane;
        if (lane < 16 && n < G_DIM) {
#pragma unroll
            for (int j = 0; j < IN_DIM; ++j) bv[j] = (_Float16)W_ih[n * IN_DIM + j];
        }
        bfr[nt] = bv;
    }
    // C preload = (b_ih + b_hh) broadcast down each output column
    v8f cb[2];
#pragma unroll
    for (int nt = 0; nt < 2; ++nt) {
        const int n = nt * 16 + l15;
        const float bb = (n < G_DIM) ? (b_ih[n] + b_hh[n]) : 0.0f;
        v8f t;
#pragma unroll
        for (int v = 0; v < 8; ++v) t[v] = bb;
        cb[nt] = t;
    }

    float h[H_DIM] = {0,0,0,0,0};
    float c[H_DIM] = {0,0,0,0,0};

    const float* obs_b = obs + (size_t)b * (T_LEN * 8);
    const float* act_b = act + (size_t)b * (T_LEN * 4);

    for (int ch = 0; ch < NCHUNK; ++ch) {
        const int t0 = ch * TC;

        // -------- stage x chunk: f32 global -> f16 LDS, 128B contiguous per lane --------
#pragma unroll
        for (int tt = 0; tt < TC; ++tt) {
            const float4* op = (const float4*)(obs_b + (size_t)(t0 + tt) * 8);
            const float4  o0 = op[0], o1 = op[1];
            const float4  a0 = *(const float4*)(act_b + (size_t)(t0 + tt) * 4);
            union { _Float16 hh[16]; uint4 q[2]; } r;
            r.hh[0]=(_Float16)o0.x; r.hh[1]=(_Float16)o0.y; r.hh[2]=(_Float16)o0.z; r.hh[3]=(_Float16)o0.w;
            r.hh[4]=(_Float16)o1.x; r.hh[5]=(_Float16)o1.y; r.hh[6]=(_Float16)o1.z; r.hh[7]=(_Float16)o1.w;
            r.hh[8]=(_Float16)a0.x; r.hh[9]=(_Float16)a0.y; r.hh[10]=(_Float16)a0.z; r.hh[11]=(_Float16)a0.w;
            r.hh[12]=(_Float16)0.f; r.hh[13]=(_Float16)0.f; r.hh[14]=(_Float16)0.f; r.hh[15]=(_Float16)0.f;
            uint4* dst = (uint4*)&xlds[(tid * TC + tt) * XSTRIDE];
            dst[0] = r.q[0]; dst[1] = r.q[1];
        }
        if (ch + 1 < NCHUNK) {
            // WGP-scope prefetch (locality 3 -> near caches, pulls into all levels).
            // Next chunk per lane: obs = 128B (two 64B lines), act = 64B (one line).
            __builtin_prefetch(obs_b + (size_t)(t0 + TC) * 8,      0, 3);
            __builtin_prefetch(obs_b + (size_t)(t0 + TC) * 8 + 16, 0, 3);
            __builtin_prefetch(act_b + (size_t)(t0 + TC) * 4,      0, 3);
        }
        __syncthreads();

        // -------- WMMA: gx[row, 0..19] = x_row @ W_ih^T + bias (K=12 zero-padded to 32) --------
        for (int mi = 0; mi < M_TILES / 2; ++mi) {
            const int m = wave * (M_TILES / 2) + mi;
            // A uses contiguous k (as B): lanes 0-15 halves0..15 <-> K=0..15; lanes 16-31 <-> K=16..31 (zero)
            union { v16h v; uint4 q[2]; } af;
            const uint4* ap = (const uint4*)&xlds[(m * 16 + l15) * XSTRIDE];
            af.q[0] = hi ? make_uint4(0u, 0u, 0u, 0u) : ap[0];
            af.q[1] = hi ? make_uint4(0u, 0u, 0u, 0u) : ap[1];
#pragma unroll
            for (int nt = 0; nt < 2; ++nt) {
                v8f d = __builtin_amdgcn_wmma_f32_16x16x32_f16(
                    false, af.v, false, bfr[nt], (short)0, cb[nt], false, false);
                const int n = nt * 16 + l15;
                if (n < G_DIM) {
#pragma unroll
                    for (int v = 0; v < 8; ++v) {      // C/D: VGPR v -> M = v (+8 for hi lanes)
                        const int row = m * 16 + v + hi * 8;
                        gxlds[row * G_DIM + n] = d[v];
                    }
                }
            }
        }
        __syncthreads();

        // -------- sequential LSTM recurrence: 1 lane = 1 batch row, TC steps --------
#pragma unroll
        for (int tt = 0; tt < TC; ++tt) {
            const float4* gp = (const float4*)&gxlds[(tid * TC + tt) * G_DIM];
            const float4 g0 = gp[0], g1 = gp[1], g2 = gp[2], g3 = gp[3], g4 = gp[4];
            float g[G_DIM] = { g0.x,g0.y,g0.z,g0.w, g1.x,g1.y,g1.z,g1.w,
                               g2.x,g2.y,g2.z,g2.w, g3.x,g3.y,g3.z,g3.w,
                               g4.x,g4.y,g4.z,g4.w };
#pragma unroll
            for (int gi = 0; gi < G_DIM; ++gi) {       // g += h @ W_hh^T  (100 fmacs, 20 indep chains)
                float acc = g[gi];
#pragma unroll
                for (int k = 0; k < H_DIM; ++k) acc = __builtin_fmaf(h[k], whh[gi * H_DIM + k], acc);
                g[gi] = acc;
            }
#pragma unroll
            for (int u = 0; u < H_DIM; ++u) {          // PyTorch gate order i,f,g,o
                const float iv = sigm(g[u]);
                const float fv = sigm(g[5 + u]);
                const float gv = fast_tanh(g[10 + u]);
                const float ov = sigm(g[15 + u]);
                c[u] = __builtin_fmaf(fv, c[u], iv * gv);
                h[u] = ov * fast_tanh(c[u]);
            }
        }
    }

    // -------- fc head on h_last --------
    float4 o;
    o.x = bfc[0]; o.y = bfc[1]; o.z = bfc[2]; o.w = bfc[3];
#pragma unroll
    for (int k = 0; k < H_DIM; ++k) {
        o.x = __builtin_fmaf(h[k], wfc[0 * H_DIM + k], o.x);
        o.y = __builtin_fmaf(h[k], wfc[1 * H_DIM + k], o.y);
        o.z = __builtin_fmaf(h[k], wfc[2 * H_DIM + k], o.z);
        o.w = __builtin_fmaf(h[k], wfc[3 * H_DIM + k], o.w);
    }
    *(float4*)(out + (size_t)b * 4) = o;
}

extern "C" void kernel_launch(void* const* d_in, const int* in_sizes, int n_in,
                              void* d_out, int out_size, void* d_ws, size_t ws_size,
                              hipStream_t stream) {
    (void)in_sizes; (void)n_in; (void)out_size; (void)d_ws; (void)ws_size;
    const float* obs  = (const float*)d_in[0];
    const float* act  = (const float*)d_in[1];
    const float* W_ih = (const float*)d_in[2];
    const float* W_hh = (const float*)d_in[3];
    const float* b_ih = (const float*)d_in[4];
    const float* b_hh = (const float*)d_in[5];
    const float* W_fc = (const float*)d_in[6];
    const float* b_fc = (const float*)d_in[7];
    float* out = (float*)d_out;

    dim3 grid(4096 / WG_BATCH);   // 64 WGs, one per 64-row batch slice
    dim3 block(WG_BATCH);         // 64 threads = 2 wave32s
    hipLaunchKernelGGL(policy_lstm_fused, grid, block, 0, stream,
                       obs, act, W_ih, W_hh, b_ih, b_hh, W_fc, b_fc, out);
}
